// VAETransformer_FCGPFA_34832184770855
// MI455X (gfx1250) — compile-verified, matrix operand order checked by reading, not run on hardware
//
#include <hip/hip_runtime.h>
#include <math.h>

// ---------- types ----------
typedef __bf16 bf16;
typedef __attribute__((ext_vector_type(16))) __bf16 v16bf;
typedef __attribute__((ext_vector_type(8)))  float  v8f;
typedef __attribute__((ext_vector_type(4)))  int    v4i;

#define NPADC   100
#define TLEN    1000
#define DMODEL  256
#define HD      32
#define BATCH   16
#define NNEUR   256
#define TTOT    1100
#define FFDIM   1024
#define MROW    (BATCH*TLEN)   // 16000

// ---------- gfx1250 async-copy support (guarded; falls back to sync copies) ----------
#if defined(__AMDGCN__) && __has_builtin(__builtin_amdgcn_global_load_async_to_lds_b128)
#define HAVE_ASYNC_LDS 1
#else
#define HAVE_ASYNC_LDS 0
#endif

__device__ __forceinline__ void wait_async_lds() {
#if HAVE_ASYNC_LDS
#if __has_builtin(__builtin_amdgcn_s_wait_asynccnt)
  __builtin_amdgcn_s_wait_asynccnt(0);
#else
  asm volatile("s_wait_asynccnt 0x0" ::: "memory");
#endif
#endif
}

__device__ __forceinline__ void copy16_g2l(const bf16* src, bf16* dst) {
#if HAVE_ASYNC_LDS
  __builtin_amdgcn_global_load_async_to_lds_b128((v4i*)src, (v4i*)dst, 0, 0);
#else
  *(uint4*)dst = *(const uint4*)src;
#endif
}

__device__ __forceinline__ bf16 f2bf(float f) {
  union { float f; unsigned u; } v; v.f = f;
  unsigned r = v.u + 0x7FFFu + ((v.u >> 16) & 1u);
  unsigned short h = (unsigned short)(r >> 16);
  bf16 o; __builtin_memcpy(&o, &h, 2); return o;
}

// ---------- small helper kernels ----------
__global__ void k_f32_to_bf16(const float* __restrict__ in, bf16* __restrict__ out, int n) {
  int i = blockIdx.x * blockDim.x + threadIdx.x;
  if (i < n) out[i] = f2bf(in[i]);
}

// A0[(b*T+t)*N + n] = Y[(b*N+n)*Ttot + NPAD + t]  (tokenizer GEMM A operand, bf16)
__global__ void k_make_tokens(const float* __restrict__ Y, bf16* __restrict__ A0) {
  int idx = blockIdx.x * blockDim.x + threadIdx.x;
  if (idx >= MROW * NNEUR) return;
  int n = idx % NNEUR; int bt = idx / NNEUR;
  int t = bt % TLEN;   int b  = bt / TLEN;
  A0[idx] = f2bf(Y[(size_t)(b*NNEUR + n)*TTOT + NPADC + t]);
}

// add sinusoidal posenc, emit f32 + bf16 activations
__global__ void k_posenc(const float* __restrict__ xin, float* __restrict__ xf, bf16* __restrict__ xb) {
  int idx = blockIdx.x * blockDim.x + threadIdx.x;
  if (idx >= MROW * DMODEL) return;
  int d = idx % DMODEL; int bt = idx / DMODEL; int t = bt % TLEN;
  int i2 = d >> 1;
  const float cc = -6.9077552789821f / 256.0f;  // -ln(1000)/D
  float freq = __expf((float)(2 * i2) * cc);
  float ang  = (float)t * freq;
  float pe   = (d & 1) ? __cosf(ang) : __sinf(ang);
  float v = xin[idx] + pe;
  xf[idx] = v; xb[idx] = f2bf(v);
}

// ---------- WMMA GEMM: C(MxN) = A(MxK,bf16) @ B(KxN,bf16) + bias ----------
// block tile 64x256, 8 waves (2x4), per-wave 32x64 (2x4 wmma 16x16x32 tiles)
// double-buffered LDS staging; A staged via async-to-LDS when available.
// mode: 0 -> f32 out, 1 -> bf16 out, 2 -> gelu(exact) -> bf16 out
__global__ __launch_bounds__(256) void k_gemm_bf16(
    const bf16* __restrict__ A, const bf16* __restrict__ B,
    const float* __restrict__ bias, void* __restrict__ C,
    int M, int N, int K, int mode) {
  __shared__ __align__(32) bf16 As[2][64][32];      // 8 KB
  __shared__ __align__(32) bf16 BsT[2][256][32];    // 32 KB (transposed: [n][k])

  int tid  = threadIdx.x;
  int lane = tid & 31;
  int wid  = tid >> 5;
  int wm   = wid >> 2;            // 0..1
  int wn   = wid & 3;             // 0..3
  int l16  = lane & 15;
  int lhalf = lane >> 4;          // 0/1 : K half selector

  int rowBase = blockIdx.x * 64;
  int colBase = blockIdx.y * 256;

  int arow = tid >> 2, akc = tid & 3;   // A tile: 64 rows x 4 chunks of 8

  v8f c[2][4] = {};

  // --- staging helpers ---
  auto stageA = [&](int k0, int p) {
    copy16_g2l(A + (size_t)(rowBase + arow) * K + k0 + akc * 8, &As[p][arow][akc * 8]);
  };
  auto stageB = [&](int k0, int p) {
    #pragma unroll
    for (int cc2 = 0; cc2 < 4; ++cc2) {
      int ch = tid + cc2 * 256;         // 0..1023
      int kk = ch >> 5;                 // 0..31
      int n0 = (ch & 31) * 8;
      uint4 raw = *(const uint4*)(B + (size_t)(k0 + kk) * N + colBase + n0);
      const bf16* pv = (const bf16*)&raw;
      #pragma unroll
      for (int i = 0; i < 8; ++i) BsT[p][n0 + i][kk] = pv[i];
      if (k0 + 64 < K) __builtin_prefetch(B + (size_t)(k0 + 64 + kk) * N + colBase + n0, 0, 1);
    }
  };

  int nt = K >> 5;
  stageA(0, 0); stageB(0, 0);

  for (int it = 0; it < nt; ++it) {
    int p = it & 1;
    wait_async_lds();
    __syncthreads();                       // stage(it) complete & prior readers of buf p^1 done
    if (it + 1 < nt) { stageA((it + 1) * 32, p ^ 1); stageB((it + 1) * 32, p ^ 1); }

    v16bf a[2], b[4];
    #pragma unroll
    for (int i = 0; i < 2; ++i)
      a[i] = *(const v16bf*)(&As[p][wm * 32 + i * 16 + l16][lhalf * 16]);
    #pragma unroll
    for (int j = 0; j < 4; ++j)
      b[j] = *(const v16bf*)(&BsT[p][wn * 64 + j * 16 + l16][lhalf * 16]);

    #pragma unroll
    for (int i = 0; i < 2; ++i)
      #pragma unroll
      for (int j = 0; j < 4; ++j)
        c[i][j] = __builtin_amdgcn_wmma_f32_16x16x32_bf16(
            false, a[i], false, b[j], (short)0, c[i][j], false, false);
  }

  // epilogue: C layout -> element (m,n): lane = n%16 + 16*(m/8), vgpr = m%8
  #pragma unroll
  for (int i = 0; i < 2; ++i) {
    #pragma unroll
    for (int j = 0; j < 4; ++j) {
      int col = colBase + wn * 64 + j * 16 + l16;
      float bv = bias ? bias[col] : 0.0f;
      #pragma unroll
      for (int v = 0; v < 8; ++v) {
        int rowL = wm * 32 + i * 16 + v + 8 * lhalf;
        size_t o = (size_t)(rowBase + rowL) * N + col;
        float val = c[i][j][v] + bv;
        if (mode == 0) {
          ((float*)C)[o] = val;
        } else if (mode == 1) {
          ((bf16*)C)[o] = f2bf(val);
        } else {
          float g = 0.5f * val * (1.0f + erff(val * 0.70710678118654752f));
          ((bf16*)C)[o] = f2bf(g);
        }
      }
    }
  }
}

// ---------- flash attention: one wave handles 16 query rows of one (b,h) ----------
// qkv rows: [q(256) k(256) v(256)] bf16, row stride 768; out: (B*T, 256) bf16
__global__ __launch_bounds__(32) void k_attn(const bf16* __restrict__ qkv, bf16* __restrict__ out) {
  int bh = blockIdx.x;                 // 0..127
  int b = bh >> 3, h = bh & 7;
  int qbase = blockIdx.y * 16;         // 0..62 tiles
  int lane = threadIdx.x;
  int l16 = lane & 15, lhalf = lane >> 4;
  const float scale = 0.17677669529663689f;  // 1/sqrt(32)

  __shared__ __align__(32) bf16 Plds[16][32];
  __shared__ __align__(32) bf16 Vlds[32][32];

  // Q fragment (A layout: lane holds row l16, K-half lhalf)
  int qrow = qbase + l16; if (qrow >= TLEN) qrow = TLEN - 1;
  v16bf qa = *(const v16bf*)(qkv + ((size_t)(b * TLEN + qrow)) * 768 + h * HD + lhalf * 16);

  v8f o0 = {}, o1 = {};
  float mrow[8], lrow[8];
  #pragma unroll
  for (int v = 0; v < 8; ++v) { mrow[v] = -1e30f; lrow[v] = 0.0f; }

  for (int kb = 0; kb < TLEN; kb += 32) {
    // --- stage V tile (32 keys x 32 hd) into LDS; lane = key row ---
    {
      int vkey = kb + lane; if (vkey >= TLEN) vkey = TLEN - 1;
      const bf16* vsrc = qkv + ((size_t)(b * TLEN + vkey)) * 768 + 512 + h * HD;
      #pragma unroll
      for (int cc2 = 0; cc2 < 4; ++cc2)
        copy16_g2l(vsrc + cc2 * 8, &Vlds[lane][cc2 * 8]);
    }

    // --- S = Q K^T ---
    v16bf kf[2];
    #pragma unroll
    for (int j = 0; j < 2; ++j) {
      int key = kb + j * 16 + l16; if (key >= TLEN) key = TLEN - 1;
      kf[j] = *(const v16bf*)(qkv + ((size_t)(b * TLEN + key)) * 768 + 256 + h * HD + lhalf * 16);
    }
    v8f z0 = {}, z1 = {};
    v8f s0 = __builtin_amdgcn_wmma_f32_16x16x32_bf16(false, qa, false, kf[0], (short)0, z0, false, false);
    v8f s1 = __builtin_amdgcn_wmma_f32_16x16x32_bf16(false, qa, false, kf[1], (short)0, z1, false, false);

    int key0 = kb + l16, key1 = kb + 16 + l16;
    float p0a[8], p1a[8];
    #pragma unroll
    for (int v = 0; v < 8; ++v) {
      float x0 = s0[v] * scale, x1 = s1[v] * scale;
      if (key0 >= TLEN) x0 = -1e30f;
      if (key1 >= TLEN) x1 = -1e30f;
      // online softmax per row (row lives across one 16-lane half at vgpr v)
      float m = fmaxf(x0, x1);
      for (int d = 1; d < 16; d <<= 1) m = fmaxf(m, __shfl_xor(m, d, 32));
      float mnew = fmaxf(mrow[v], m);
      float alpha = __expf(mrow[v] - mnew);
      mrow[v] = mnew;
      float p0 = __expf(x0 - mnew), p1 = __expf(x1 - mnew);
      p0a[v] = p0; p1a[v] = p1;
      float s = p0 + p1;
      for (int d = 1; d < 16; d <<= 1) s += __shfl_xor(s, d, 32);
      lrow[v] = lrow[v] * alpha + s;
      o0[v] *= alpha; o1[v] *= alpha;
    }

    // relayout P: C-layout -> A-layout via LDS
    __syncthreads();
    #pragma unroll
    for (int v = 0; v < 8; ++v) {
      int r = v + 8 * lhalf;
      Plds[r][l16]      = f2bf(p0a[v]);
      Plds[r][16 + l16] = f2bf(p1a[v]);
    }
    wait_async_lds();
    __syncthreads();
    v16bf pa = *(const v16bf*)(&Plds[l16][lhalf * 16]);

    // V fragments from LDS (B layout: lane = hd column, contiguous keys)
    v16bf vf[2];
    #pragma unroll
    for (int j = 0; j < 2; ++j) {
      int hc = j * 16 + l16;
      #pragma unroll
      for (int i = 0; i < 16; ++i)
        vf[j][i] = Vlds[lhalf * 16 + i][hc];
    }
    o0 = __builtin_amdgcn_wmma_f32_16x16x32_bf16(false, pa, false, vf[0], (short)0, o0, false, false);
    o1 = __builtin_amdgcn_wmma_f32_16x16x32_bf16(false, pa, false, vf[1], (short)0, o1, false, false);
    __syncthreads();
  }

  #pragma unroll
  for (int v = 0; v < 8; ++v) {
    int r = v + 8 * lhalf;
    int t = qbase + r;
    if (t < TLEN) {
      float inv = 1.0f / lrow[v];
      size_t o = ((size_t)(b * TLEN + t)) * DMODEL + h * HD;
      out[o + l16]      = f2bf(o0[v] * inv);
      out[o + 16 + l16] = f2bf(o1[v] * inv);
    }
  }
}

// ---------- residual + layernorm -> f32 + bf16 ----------
__global__ __launch_bounds__(256) void k_residual_ln(
    const float* xin, const float* __restrict__ delta,
    const float* __restrict__ g, const float* __restrict__ bta,
    float* xf, bf16* __restrict__ xb) {
  int row = blockIdx.x, d = threadIdx.x;
  __shared__ float red[8];
  size_t o = (size_t)row * DMODEL + d;
  float v = xin[o] + delta[o];
  float s = v;
  for (int m = 1; m < 32; m <<= 1) s += __shfl_xor(s, m, 32);
  int wid = d >> 5, lane = d & 31;
  if (lane == 0) red[wid] = s;
  __syncthreads();
  float tot = 0.f;
  #pragma unroll
  for (int i = 0; i < 8; ++i) tot += red[i];
  float mean = tot / 256.0f;
  __syncthreads();
  float dv = v - mean;
  float s2 = dv * dv;
  for (int m = 1; m < 32; m <<= 1) s2 += __shfl_xor(s2, m, 32);
  if (lane == 0) red[wid] = s2;
  __syncthreads();
  float tot2 = 0.f;
  #pragma unroll
  for (int i = 0; i < 8; ++i) tot2 += red[i];
  float var = tot2 / 256.0f;
  float y = dv * rsqrtf(var + 1e-5f) * g[d] + bta[d];
  xf[o] = y; xb[o] = f2bf(y);
}

// ---------- mean pool over T ----------
__global__ void k_pool(const float* __restrict__ x, float* __restrict__ pooled) {
  int b = blockIdx.x, d = threadIdx.x;
  float s = 0.f;
  for (int t = 0; t < TLEN; ++t) s += x[((size_t)(b * TLEN + t)) * DMODEL + d];
  pooled[b * DMODEL + d] = s * (1.0f / (float)TLEN);
}

// ---------- tiny scalar GEMM for the decoder head ----------
__global__ void k_small_gemm(const float* __restrict__ A, int lda,
                             const float* __restrict__ W, const float* __restrict__ bias,
                             float* __restrict__ C, int M, int N, int K, int act) {
  int i = blockIdx.x * blockDim.x + threadIdx.x;
  if (i >= M * N) return;
  int m = i / N, n = i % N;
  float s = bias ? bias[n] : 0.f;
  for (int k = 0; k < K; ++k) s += A[m * lda + k] * W[k * N + n];
  if (act == 1) s = fmaxf(s, 0.f);
  C[m * N + n] = s;
}

// ---------- sti_lat[b,f,t] = sum_k coeffs[b,f,k]*basis[t,k] ----------
__global__ void k_stilat(const float* __restrict__ coeffs, const float* __restrict__ basis,
                         float* __restrict__ lat) {
  int bf = blockIdx.x;               // b*4+f
  int b = bf >> 2, f = bf & 3;
  for (int t = threadIdx.x; t < TLEN; t += blockDim.x) {
    float s = 0.f;
    #pragma unroll
    for (int k = 0; k < 30; ++k) s += coeffs[b * 120 + f * 30 + k] * basis[t * 30 + k];
    lat[(size_t)bf * TLEN + t] = s;
  }
}

// ---------- final: sti + coupling conv + self conv ----------
__global__ __launch_bounds__(256) void k_final(
    const float* __restrict__ Y, const float* __restrict__ lat,
    const float* __restrict__ readout_W, const float* __restrict__ coupling_W,
    const float* __restrict__ coupling_basis, const float* __restrict__ self_kernel,
    float* __restrict__ outp) {
  int bn = blockIdx.x;                // b*256+n
  int b = bn >> 8, n = bn & 255;
  __shared__ float cbS[20][4];
  __shared__ float skS[20];
  if (threadIdx.x < 80) { int tau = threadIdx.x >> 2, s = threadIdx.x & 3; cbS[tau][s] = coupling_basis[tau * 4 + s]; }
  if (threadIdx.x < 20) skS[threadIdx.x] = self_kernel[threadIdx.x * 256 + n];
  __syncthreads();
  float cw0 = coupling_W[n * 4 + 0], cw1 = coupling_W[n * 4 + 1];
  float cw2 = coupling_W[n * 4 + 2], cw3 = coupling_W[n * 4 + 3];
  float rw0 = readout_W[n * 4 + 0],  rw1 = readout_W[n * 4 + 1];
  float rw2 = readout_W[n * 4 + 2],  rw3 = readout_W[n * 4 + 3];
  const float* Yb = Y + ((size_t)bn) * TTOT + NPADC;
  for (int t = threadIdx.x; t < TLEN; t += 256) {
    float g0 = 0, g1 = 0, g2 = 0, g3 = 0, slf = 0;
    #pragma unroll
    for (int tau = 1; tau <= 20; ++tau) {
      float y = Yb[t - tau];
      g0 += y * cbS[tau - 1][0]; g1 += y * cbS[tau - 1][1];
      g2 += y * cbS[tau - 1][2]; g3 += y * cbS[tau - 1][3];
      slf += y * skS[tau - 1];
    }
    if (fabsf(g0) < 1e-5f) g0 = 0.f;
    if (fabsf(g1) < 1e-5f) g1 = 0.f;
    if (fabsf(g2) < 1e-5f) g2 = 0.f;
    if (fabsf(g3) < 1e-5f) g3 = 0.f;
    if (fabsf(slf) < 1e-5f) slf = 0.f;
    float cpl = cw0 * g0 + cw1 * g1 + cw2 * g2 + cw3 * g3;
    float sti = rw0 * lat[(size_t)(b * 4 + 0) * TLEN + t] + rw1 * lat[(size_t)(b * 4 + 1) * TLEN + t]
              + rw2 * lat[(size_t)(b * 4 + 2) * TLEN + t] + rw3 * lat[(size_t)(b * 4 + 3) * TLEN + t];
    outp[(size_t)bn * TLEN + t] = sti + cpl + slf;
  }
}

// ---------- host orchestration ----------
extern "C" void kernel_launch(void* const* d_in, const int* in_sizes, int n_in,
                              void* d_out, int out_size, void* d_ws, size_t ws_size,
                              hipStream_t stream) {
  (void)in_sizes; (void)n_in; (void)out_size; (void)ws_size;
  const float* Y        = (const float*)d_in[0];
  const float* token_W  = (const float*)d_in[1];
  const float* token_b  = (const float*)d_in[2];
  const float* Wqkv     = (const float*)d_in[3];
  const float* bqkv     = (const float*)d_in[4];
  const float* Wo       = (const float*)d_in[5];
  const float* bo       = (const float*)d_in[6];
  const float* ln1_g    = (const float*)d_in[7];
  const float* ln1_b    = (const float*)d_in[8];
  const float* W1       = (const float*)d_in[9];
  const float* b1       = (const float*)d_in[10];
  const float* W2       = (const float*)d_in[11];
  const float* b2       = (const float*)d_in[12];
  const float* ln2_g    = (const float*)d_in[13];
  const float* ln2_b    = (const float*)d_in[14];
  const float* toL_W    = (const float*)d_in[15];
  const float* toL_b    = (const float*)d_in[16];
  const float* dec_W1   = (const float*)d_in[17];
  const float* dec_b1   = (const float*)d_in[18];
  const float* dec_W2   = (const float*)d_in[19];
  const float* dec_b2   = (const float*)d_in[20];
  const float* stim     = (const float*)d_in[21];
  const float* readoutW = (const float*)d_in[22];
  const float* couplW   = (const float*)d_in[23];
  const float* couplB   = (const float*)d_in[24];
  const float* selfK    = (const float*)d_in[25];

  char* ws = (char*)d_ws;
  size_t off = 0;
  auto alloc = [&](size_t bytes) -> void* {
    off = (off + 255) & ~(size_t)255;
    void* p = ws + off;
    off += bytes;
    return p;
  };

  float* xf32   = (float*)alloc((size_t)MROW * DMODEL * 4);
  bf16*  xbf    = (bf16*) alloc((size_t)MROW * DMODEL * 2);
  float* tmpf   = (float*)alloc((size_t)MROW * DMODEL * 4);
  bf16*  qkvbf  = (bf16*) alloc((size_t)MROW * 768 * 2);
  bf16*  attnbf = (bf16*) alloc((size_t)MROW * DMODEL * 2);   // also reused as token-A0
  bf16*  ff1bf  = (bf16*) alloc((size_t)MROW * FFDIM * 2);
  bf16*  tokWbf = (bf16*) alloc((size_t)256 * 256 * 2);
  bf16*  Wqkvbf = (bf16*) alloc((size_t)4 * 256 * 768 * 2);
  bf16*  Wobf   = (bf16*) alloc((size_t)4 * 256 * 256 * 2);
  bf16*  W1bf   = (bf16*) alloc((size_t)4 * 256 * 1024 * 2);
  bf16*  W2bf   = (bf16*) alloc((size_t)4 * 1024 * 256 * 2);
  float* pooled = (float*)alloc(16 * 256 * 4);
  float* hbuf   = (float*)alloc(16 * 128 * 4);
  float* c1buf  = (float*)alloc(16 * 128 * 4);
  float* coeffb = (float*)alloc(16 * 120 * 4);
  float* latb   = (float*)alloc(64 * 1000 * 4);

  // weight conversions f32 -> bf16
  k_f32_to_bf16<<<(65536 + 255) / 256, 256, 0, stream>>>(token_W, tokWbf, 65536);
  k_f32_to_bf16<<<(786432 + 255) / 256, 256, 0, stream>>>(Wqkv, Wqkvbf, 786432);
  k_f32_to_bf16<<<(262144 + 255) / 256, 256, 0, stream>>>(Wo, Wobf, 262144);
  k_f32_to_bf16<<<(1048576 + 255) / 256, 256, 0, stream>>>(W1, W1bf, 1048576);
  k_f32_to_bf16<<<(1048576 + 255) / 256, 256, 0, stream>>>(W2, W2bf, 1048576);

  // tokenizer: A0 (reuse attnbf) -> GEMM -> +posenc
  k_make_tokens<<<(MROW * NNEUR) / 256, 256, 0, stream>>>(Y, attnbf);
  {
    dim3 g(MROW / 64, DMODEL / 256);
    k_gemm_bf16<<<g, 256, 0, stream>>>(attnbf, tokWbf, token_b, tmpf, MROW, DMODEL, 256, 0);
  }
  k_posenc<<<(MROW * DMODEL) / 256, 256, 0, stream>>>(tmpf, xf32, xbf);

  // encoder layers
  for (int l = 0; l < 4; ++l) {
    { dim3 g(MROW / 64, 768 / 256);
      k_gemm_bf16<<<g, 256, 0, stream>>>(xbf, Wqkvbf + (size_t)l * 256 * 768, bqkv + l * 768,
                                         qkvbf, MROW, 768, 256, 1); }
    { dim3 g(BATCH * 8, (TLEN + 15) / 16);
      k_attn<<<g, 32, 0, stream>>>(qkvbf, attnbf); }
    { dim3 g(MROW / 64, DMODEL / 256);
      k_gemm_bf16<<<g, 256, 0, stream>>>(attnbf, Wobf + (size_t)l * 65536, bo + l * 256,
                                         tmpf, MROW, DMODEL, 256, 0); }
    k_residual_ln<<<MROW, 256, 0, stream>>>(xf32, tmpf, ln1_g + l * 256, ln1_b + l * 256, xf32, xbf);
    { dim3 g(MROW / 64, FFDIM / 256);
      k_gemm_bf16<<<g, 256, 0, stream>>>(xbf, W1bf + (size_t)l * 262144, b1 + l * 1024,
                                         ff1bf, MROW, FFDIM, 256, 2); }
    { dim3 g(MROW / 64, DMODEL / 256);
      k_gemm_bf16<<<g, 256, 0, stream>>>(ff1bf, W2bf + (size_t)l * 262144, b2 + l * 256,
                                         tmpf, MROW, DMODEL, FFDIM, 0); }
    k_residual_ln<<<MROW, 256, 0, stream>>>(xf32, tmpf, ln2_g + l * 256, ln2_b + l * 256, xf32, xbf);
  }

  // decoder head
  k_pool<<<BATCH, 256, 0, stream>>>(xf32, pooled);
  k_small_gemm<<<(16 * 128 + 255) / 256, 256, 0, stream>>>(pooled, 256, toL_W, toL_b, hbuf, 16, 128, 256, 0);
  k_small_gemm<<<(16 * 128 + 255) / 256, 256, 0, stream>>>(hbuf, 128, dec_W1, dec_b1, c1buf, 16, 128, 64, 1);
  k_small_gemm<<<(16 * 120 + 255) / 256, 256, 0, stream>>>(c1buf, 128, dec_W2, dec_b2, coeffb, 16, 120, 128, 0);
  k_stilat<<<64, 256, 0, stream>>>(coeffb, stim, latb);

  // final combine: stimulus + coupling conv + self conv
  k_final<<<BATCH * NNEUR, 256, 0, stream>>>(Y, latb, readoutW, couplW, couplB, selfK, (float*)d_out);
}
